// DPMPModel_47218870452460
// MI455X (gfx1250) — matrix operand order repaired and checked by reading, not compiled, not run on hardware
//
#include <hip/hip_runtime.h>
#include <hip/hip_bf16.h>
#include <math.h>

// ---------------------------------------------------------------------------
// DPMP forward for MI455X (gfx1250, wave32, WMMA).
// Fused pipeline: no (N,M,32) tensor ever touches HBM; pair geometry is
// recomputed (cheap VALU) in every consuming kernel. The dominant GEMM work
// (two 32x32 MP layers over ~1M pairs) runs on v_wmma_f32_16x16x32_f16, and
// the ~100M tanh evaluations use the CDNA5 hardware V_TANH_F32 so they issue
// as single TRANS ops instead of branchy libm expansions.
//
// Input leaf order (jax pytree, dict keys sorted):
//   0 coord_3N(576) 1 box_33(9) 2 Tbias(32)
//   3..34  embedIJ: (i,j,k)*4 leaves [W0,b0,W1,b1]
//   35..58 embedMP: (i,j)*6 leaves   [b0,W1,b1,W2,b2,dt]
//   59..82 embedR : (i,j)*6 leaves   [W0,b0,W1,b1,W2,b2]
//   83..94 fit    : i*6 leaves       [W0,b0,W1,b1,W2,b2]
//   95,96 linear1  97,98 linear3  99,100 norm1  101,102 norm2
// ---------------------------------------------------------------------------

typedef __attribute__((ext_vector_type(16))) _Float16 v16h;
typedef __attribute__((ext_vector_type(8)))  float    v8f;

#define NATOM 192
#define NHALF 96
#define LIMG  27
#define MCOL  5184   /* 192*27 */
#define MHALF 2592   /* 96*27  */
#define RCUT  6.0f
#define NORMI (1.0f/40.0f)

__device__ __constant__ float SRM_c[2] = {0.05f, 0.06f};
__device__ __constant__ float SRS_c[2] = {0.10f, 0.12f};

// workspace layout (floats)
#define WS_T4 0                     /* [192][4][32]       = 24576 */
#define WS_F1 24576                 /* [c*2+j][192][32]   = 49152 */
#define WS_F3 73728                 /* [c*2+j][3][192][32]=147456 */
#define WS_TX 221184                /* [192][10][32]      = 61440 */
#define WS_EA 282624                /* [192]                      */

// --------------------- CDNA5 hardware-transcendental helpers ---------------
__device__ inline float htanh(float x) {
#if __has_builtin(__builtin_amdgcn_tanhf)
  return __builtin_amdgcn_tanhf(x);        // V_TANH_F32 (gfx1250 tanh-insts)
#else
  return tanhf(x);
#endif
}

__device__ inline float hrcp(float x) {
#if __has_builtin(__builtin_amdgcn_rcpf)
  return __builtin_amdgcn_rcpf(x);         // V_RCP_F32
#else
  return 1.0f / x;
#endif
}

// cos(pi*r/RCUT): V_COS_F32 computes cos(2*pi*x) -> feed r/(2*RCUT)
__device__ inline float hcos_pi_over_rcut(float r) {
#if __has_builtin(__builtin_amdgcn_cosf)
  return __builtin_amdgcn_cosf(r * (1.0f / (2.0f * RCUT)));
#else
  return cosf(3.14159265358979f * r / RCUT);
#endif
}

// ------------------------- pair geometry helper ----------------------------
struct PairGeo { float sc, srn, xn0, xn1, xn2; };

__device__ inline PairGeo pair_geo(int m, int ti, const float* __restrict__ coord,
                                   const float* __restrict__ bx, const float* __restrict__ cn) {
  int ja = m / LIMG, s = m - ja * LIMG;
  float sx = (float)(s / 9 - 1);
  float sy = (float)((s / 3) % 3 - 1);
  float sz = (float)(s % 3 - 1);
  float dx[3];
#pragma unroll
  for (int d = 0; d < 3; ++d)
    dx[d] = coord[d * NATOM + ja] + sx * bx[0 * 3 + d] + sy * bx[1 * 3 + d] + sz * bx[2 * 3 + d] - cn[d];
  float d2 = dx[0] * dx[0] + dx[1] * dx[1] + dx[2] * dx[2];
  float r = sqrtf(fmaxf(d2, 1e-30f));
  bool msk = (r > 2e-15f) && (r < RCUT);
  float sr = msk ? 0.5f * (hcos_pi_over_rcut(r) + 1.0f) * hrcp(r) : 0.0f;
  float rin = (r > 2e-15f) ? hrcp(r + 1e-16f) : 0.0f;
  PairGeo g;
  g.sc  = (sr - SRM_c[ti]) / SRS_c[ti];
  g.srn = sr / SRS_c[ti];
  g.xn0 = dx[0] * rin; g.xn1 = dx[1] * rin; g.xn2 = dx[2] * rin;
  return g;
}

// ------------------------- WMMA fragment helpers ---------------------------
// A (16x32 f16) fragment from LDS row-major stage [p][e]:
//   lane<16 : row=lane,    K in {0..7, 16..23}
//   lane>=16: row=lane-16, K in {8..15, 24..31}
__device__ inline v16h a_frag(const _Float16 (*Ast)[32], int lane) {
  union { v16h v; _Float16 h[16]; } u;
  int row = lane & 15;
  int kb  = (lane >> 4) * 8;
#pragma unroll
  for (int q = 0; q < 8; ++q) { u.h[q] = Ast[row][kb + q]; u.h[8 + q] = Ast[row][16 + kb + q]; }
  return u.v;
}

// B (32x16 f16) fragment from row-major f32 W[32][32], N-half at eoff:
//   lanes 0-15: K in {0..7, 16..23} col=lane ; lanes 16-31: K in {8..15, 24..31} col=lane-16
__device__ inline v16h b_frag(const float* __restrict__ W, int lane, int eoff) {
  union { v16h v; _Float16 h[16]; } u;
  int e  = eoff + (lane & 15);
  int kb = (lane >> 4) * 8;
#pragma unroll
  for (int q = 0; q < 8; ++q) { u.h[q] = (_Float16)W[(kb + q) * 32 + e]; u.h[8 + q] = (_Float16)W[(16 + kb + q) * 32 + e]; }
  return u.v;
}

// ===========================================================================
// Kernel 1: T_N4B[n][x][b] = (1/40) sum_m R4[x](n,m) * embedIJ(n,m)[b]
// one block per atom row; LDS tiles of 64 pairs
// ===========================================================================
struct K1Args {
  const float* coord; const float* box;
  const float* ijW0[8]; const float* ijb0[8]; const float* ijW1[8]; const float* ijb1[8];
  float* T4;
};

__global__ __launch_bounds__(128) void k1_t4(K1Args A) {
  const int n = blockIdx.x, tid = threadIdx.x;
  const int ti = (n >= NHALF) ? 1 : 0;
  __shared__ float wts[4][160];     // (j,k) nets, row type ti
  __shared__ float pb[64][6];       // sc, srn, xn0..2
  __shared__ float eb[64][32];      // embedIJ outputs
  __shared__ float bx[9], cn[3];

  for (int t = tid; t < 4 * 160; t += 128) {
    int net = t / 160, o = t - net * 160;
    int jj = net >> 1, kk = net & 1;
    int gi = (ti * 2 + jj) * 2 + kk;
    float v;
    if (o < 8)        v = A.ijW0[gi][o];
    else if (o < 16)  v = A.ijb0[gi][o - 8];
    else if (o < 144) v = A.ijW1[gi][o - 16];
    else              v = A.ijb1[gi][o - 144];
    wts[net][o] = v;
  }
  if (tid < 9) bx[tid] = A.box[tid];
  if (tid < 3) cn[tid] = A.coord[tid * NATOM + n];
  const int xo = tid >> 5, bo = tid & 31;
  float acc = 0.0f;
  __syncthreads();

  for (int t = 0; t < MCOL / 64; ++t) {
    if (tid < 64) {
      PairGeo g = pair_geo(t * 64 + tid, ti, A.coord, bx, cn);
      pb[tid][0] = g.sc; pb[tid][1] = g.srn;
      pb[tid][2] = g.xn0; pb[tid][3] = g.xn1; pb[tid][4] = g.xn2;
    }
    __syncthreads();
    {
      int pp = tid >> 1, kk = tid & 1;
      int m = t * 64 + pp;
      int tj = (m >= MHALF) ? 1 : 0;
      const float* W = wts[tj * 2 + kk];
      float sc = pb[pp][0];
      float h8[8];
#pragma unroll
      for (int c = 0; c < 8; ++c) h8[c] = htanh(sc * W[c] + W[8 + c]);
#pragma unroll
      for (int c = 0; c < 16; ++c) {
        float z = W[144 + c];
#pragma unroll
        for (int d = 0; d < 8; ++d) z += h8[d] * W[16 + d * 16 + c];
        eb[pp][kk * 16 + c] = h8[c & 7] + htanh(z);   // concat([h,h]) + tanh(y)
      }
    }
    __syncthreads();
    for (int p2 = 0; p2 < 64; ++p2) {
      float srn = pb[p2][1];
      float rx = (xo == 0) ? srn : 1.7320508075688772f * srn * pb[p2][1 + xo];
      acc = fmaf(rx, eb[p2][bo], acc);
    }
    __syncthreads();
  }
  A.T4[(n * 4 + xo) * 32 + bo] = acc * NORMI;
}

// ===========================================================================
// Kernel 2: per-atom T_2ND / F1 / F3 projections
// ===========================================================================
struct K2Args {
  const float* T4;
  const float* L1[2]; const float* L3[2]; const float* N1[2]; const float* N2[2];
  float* F1; float* F3;
};

__global__ __launch_bounds__(32) void k2_proj(K2Args A) {
  const int n = blockIdx.x, tid = threadIdx.x;
  const int ti = (n >= NHALF) ? 1 : 0;
  __shared__ float Tl[4][32];
  __shared__ float T2ND[2][64];
  for (int t = tid; t < 128; t += 32) Tl[t >> 5][t & 31] = A.T4[n * 128 + t];
  __syncthreads();
  for (int t = tid; t < 128; t += 32) {
    int c = t >> 6, d = t & 63, d1 = d >> 4, d2 = d & 15;
    float z = Tl[0][c * 16 + d2] * Tl[0][c * 16 + d1];
#pragma unroll
    for (int x = 0; x < 3; ++x) z += Tl[1 + x][c * 16 + d2] * Tl[1 + x][c * 16 + d1];
    T2ND[c][d] = z;
  }
  __syncthreads();
  const int e = tid;
  for (int c = 0; c < 2; ++c)
    for (int j = 0; j < 2; ++j) {
      int cj = c * 2 + j;
      float n1 = A.N1[ti][cj]; n1 *= n1;
      float n2 = A.N2[ti][cj]; n2 *= n2;
      float z = 0.0f;
      for (int d = 0; d < 64; ++d) z = fmaf(T2ND[c][d], A.L1[ti][(cj * 64 + d) * 32 + e], z);
      A.F1[(cj * NATOM + n) * 32 + e] = z * n1;
#pragma unroll
      for (int x = 0; x < 3; ++x) {
        float z3 = 0.0f;
        for (int cc = 0; cc < 16; ++cc)
          z3 = fmaf(Tl[1 + x][c * 16 + cc], A.L3[ti][(cj * 16 + cc) * 32 + e], z3);
        A.F3[((cj * 3 + x) * NATOM + n) * 32 + e] = z3 * n2;
      }
    }
}

// ===========================================================================
// Kernel 3: heavy fused MP pass + T_NXC reduction. WMMA 16x16x32 f16.
// one block per atom row (2 waves); each wave owns a 16-pair tile.
// ===========================================================================
struct K3Args {
  const float* coord; const float* box;
  const float* F1; const float* F3; float* TX;
  const float* erW0[4]; const float* erb0[4]; const float* erW1[4];
  const float* erb1[4]; const float* erW2[4]; const float* erb2[4];
  const float* mpb0[4]; const float* mpW1[4]; const float* mpb1[4];
  const float* mpW2[4]; const float* mpb2[4]; const float* mpdt[4];
};

__global__ __launch_bounds__(64) void k3_mp(K3Args A) {
  const int n = blockIdx.x, tid = threadIdx.x;
  const int ti = (n >= NHALF) ? 1 : 0;
  const int w = tid >> 5, lane = tid & 31;

  __shared__ float   eR[2][704];          // embedR weights per column type j
  __shared__ float   mb[2][4][32];        // MP b0,b1,b2,dt per j
  __shared__ float   FIrow[2][32];
  __shared__ float   F3I[2][3][32];
  __shared__ _Float16 Ast[2][16][32];     // WMMA A stage per wave
  __shared__ float   Hbuf[2][16][32];     // f32 activations per wave
  __shared__ float   RXs[2][10][16];      // R_X per wave-tile
  __shared__ float   accR[2][10][32];
  __shared__ float   bx[9], cn[3];

  for (int j = 0; j < 2; ++j) {
    int ij = ti * 2 + j;
    for (int t = tid; t < 8;   t += 64) { eR[j][t] = A.erW0[ij][t]; eR[j][8 + t] = A.erb0[ij][t]; }
    for (int t = tid; t < 128; t += 64) eR[j][16 + t] = A.erW1[ij][t];
    for (int t = tid; t < 16;  t += 64) eR[j][144 + t] = A.erb1[ij][t];
    for (int t = tid; t < 512; t += 64) eR[j][160 + t] = A.erW2[ij][t];
    for (int t = tid; t < 32;  t += 64) {
      eR[j][672 + t] = A.erb2[ij][t];
      mb[j][0][t] = A.mpb0[ij][t]; mb[j][1][t] = A.mpb1[ij][t];
      mb[j][2][t] = A.mpb2[ij][t]; mb[j][3][t] = A.mpdt[ij][t];
      FIrow[j][t] = A.F1[(j * NATOM + n) * 32 + t];             // c=0
    }
    for (int t = tid; t < 96; t += 64) {
      int x = t >> 5, e = t & 31;
      F3I[j][x][e] = A.F3[((j * 3 + x) * NATOM + n) * 32 + e];  // c=0
    }
  }
  if (tid < 9) bx[tid] = A.box[tid];
  if (tid < 3) cn[tid] = A.coord[tid * NATOM + n];
  float acc[10];
#pragma unroll
  for (int x = 0; x < 10; ++x) acc[x] = 0.0f;
  __syncthreads();

  for (int j = 0; j < 2; ++j) {
    const int ij = ti * 2 + j;
    // weight fragments for this (ti,j), kept in VGPRs across the tile loop
    v16h w1l = b_frag(A.mpW1[ij], lane, 0);
    v16h w1h = b_frag(A.mpW1[ij], lane, 16);
    v16h w2l = b_frag(A.mpW2[ij], lane, 0);
    v16h w2h = b_frag(A.mpW2[ij], lane, 16);

    for (int st = 0; st < MHALF / 32; ++st) {
      const int p  = lane & 15;
      const int hf = lane >> 4;                 // feature half this lane fills
      const int m  = j * MHALF + st * 32 + w * 16 + p;
      const int ja = m / LIMG;

      PairGeo g = pair_geo(m, ti, A.coord, bx, cn);
      float xn[3] = {g.xn0, g.xn1, g.xn2};
      float R3[3];
#pragma unroll
      for (int d = 0; d < 3; ++d) R3[d] = 1.7320508075688772f * g.srn * xn[d];
      if (hf == 0) {
        RXs[w][0][p] = g.srn;
        float msq = (xn[0] * xn[0] + xn[1] * xn[1] + xn[2] * xn[2]) * (1.0f / 3.0f);
#pragma unroll
        for (int d = 0; d < 3; ++d) {
          RXs[w][1 + d][p] = R3[d];
          RXs[w][4 + d][p] = 3.0f * g.srn * (xn[d] * xn[d] - msq);
          RXs[w][7 + d][p] = 4.242640687119285f * g.srn * xn[d] * xn[(d + 1) % 3];
        }
      }
      // embedR MLP (1 -> 8 -> 16(res-concat) -> 32 linear)
      float h8[8];
#pragma unroll
      for (int c = 0; c < 8; ++c) h8[c] = htanh(g.sc * eR[j][c] + eR[j][8 + c]);
      float h16[16];
#pragma unroll
      for (int c = 0; c < 16; ++c) {
        float z = eR[j][144 + c];
#pragma unroll
        for (int d = 0; d < 8; ++d) z += h8[d] * eR[j][16 + d * 16 + c];
        h16[c] = h8[c & 7] + htanh(z);
      }
      // F = FI + FJ + FI3 + FJ3 + embedR ; h0 = tanh(F + b0)
      const float* FJ  = A.F1 + ((2 + ti) * NATOM + ja) * 32;   // c=1, idx=ti
      const int eb0 = hf * 16;
#pragma unroll
      for (int q = 0; q < 16; ++q) {
        int e = eb0 + q;
        float f = FIrow[j][e] + FJ[e];
#pragma unroll
        for (int x = 0; x < 3; ++x) {
          f = fmaf(R3[x], F3I[j][x][e], f);
          f = fmaf(R3[x], A.F3[(((2 + ti) * 3 + x) * NATOM + ja) * 32 + e], f);
        }
        float er = eR[j][672 + e];
#pragma unroll
        for (int c = 0; c < 16; ++c) er = fmaf(h16[c], eR[j][160 + c * 32 + e], er);
        float h0 = htanh(f + er + mb[j][0][e]);
        Hbuf[w][p][e] = h0;
        Ast[w][p][e]  = (_Float16)h0;
      }
      __syncthreads();                                   // h0 staged

      // ---- MP layer 1: Y = h0 @ W1 (two 16x16x32 WMMAs) ----
      v16h af = a_frag(Ast[w], lane);
      v8f y0 = {}, y1 = {};
      y0 = __builtin_amdgcn_wmma_f32_16x16x32_f16(false, af, false, w1l, (short)0, y0, false, false);
      y1 = __builtin_amdgcn_wmma_f32_16x16x32_f16(false, af, false, w1h, (short)0, y1, false, false);
      __syncthreads();                                   // A-frag reads done
      {
        int col = lane & 15, rb = (lane >> 4) * 8;
        union { v8f v; float f[8]; } u0, u1; u0.v = y0; u1.v = y1;
#pragma unroll
        for (int v = 0; v < 8; ++v) {
          int row = rb + v;
          float ha = Hbuf[w][row][col]      + htanh(u0.f[v] + mb[j][1][col]);
          float hb = Hbuf[w][row][16 + col] + htanh(u1.f[v] + mb[j][1][16 + col]);
          Hbuf[w][row][col] = ha;      Ast[w][row][col] = (_Float16)ha;
          Hbuf[w][row][16 + col] = hb; Ast[w][row][16 + col] = (_Float16)hb;
        }
      }
      __syncthreads();                                   // h1 staged

      // ---- MP layer 2: Y2 = h1 @ W2 ; h2 = h1 + tanh(Y2+b2)*dt ----
      v16h af2 = a_frag(Ast[w], lane);
      v8f z0 = {}, z1 = {};
      z0 = __builtin_amdgcn_wmma_f32_16x16x32_f16(false, af2, false, w2l, (short)0, z0, false, false);
      z1 = __builtin_amdgcn_wmma_f32_16x16x32_f16(false, af2, false, w2h, (short)0, z1, false, false);
      {
        int col = lane & 15, rb = (lane >> 4) * 8;
        union { v8f v; float f[8]; } u0, u1; u0.v = z0; u1.v = z1;
#pragma unroll
        for (int v = 0; v < 8; ++v) {
          int row = rb + v;
          Hbuf[w][row][col]      += htanh(u0.f[v] + mb[j][2][col])      * mb[j][3][col];
          Hbuf[w][row][16 + col] += htanh(u1.f[v] + mb[j][2][16 + col]) * mb[j][3][16 + col];
        }
      }
      __syncthreads();                                   // h2 visible

      // ---- fused T_NXC accumulation: lane owns feature column b=lane ----
      for (int p2 = 0; p2 < 16; ++p2) {
        float hv = Hbuf[w][p2][lane];
#pragma unroll
        for (int x = 0; x < 10; ++x) acc[x] = fmaf(RXs[w][x][p2], hv, acc[x]);
      }
      __syncthreads();                                   // staging reusable
    }
  }
#pragma unroll
  for (int x = 0; x < 10; ++x) accR[w][x][lane] = acc[x];
  __syncthreads();
  for (int t = tid; t < 320; t += 64) {
    int x = t >> 5, b = t & 31;
    A.TX[(n * 10 + x) * 32 + b] = (accR[0][x][b] + accR[1][x][b]) * NORMI;
  }
}

// ===========================================================================
// Kernel 4: symmetry features G + fit net -> per-atom energy
// ===========================================================================
struct K4Args {
  const float* TX; const float* Tbias;
  const float* fW0[2]; const float* fb0[2]; const float* fW1[2];
  const float* fb1[2]; const float* fW2[2]; const float* fb2[2];
  float* EA;
};

__global__ __launch_bounds__(64) void k4_fit(K4Args A) {
  const int n = blockIdx.x, tid = threadIdx.x;
  const int ti = (n >= NHALF) ? 1 : 0;
  __shared__ float TX[10][32], TNC[32], G[512], G11a[6][4], y0s[64], y1s[64], red[64];
  for (int t = tid; t < 320; t += 64) TX[t >> 5][t & 31] = A.TX[n * 320 + t];
  __syncthreads();
  if (tid < 32) TNC[tid] = TX[0][tid] + A.Tbias[tid];
  if (tid < 24) {
    int s = tid >> 2, a = tid & 3;
    if (s < 3) G11a[s][a] = TX[1 + s][8 + a] * TX[1 + s][8 + a];
    else {
      int d = s - 3;
      G11a[s][a] = 1.4142135623730951f * TX[1 + d][8 + a] * TX[1 + ((d + 1) % 3)][8 + a];
    }
  }
  __syncthreads();
  for (int gidx = tid; gidx < 512; gidx += 64) {
    int row = gidx >> 5, b = gidx & 31, a = row & 3, sec = row >> 2;
    float z = 0.0f;
    if (sec == 0) z = TNC[b] * TNC[a];
    else if (sec == 1) {
#pragma unroll
      for (int d = 0; d < 3; ++d) z += TX[1 + d][4 + a] * TX[1 + d][b];
    } else if (sec == 2) {
#pragma unroll
      for (int s = 0; s < 6; ++s) z += G11a[s][a] * TX[4 + s][b];
    } else {
#pragma unroll
      for (int s = 0; s < 6; ++s) z += TX[4 + s][12 + a] * TX[4 + s][b];
    }
    G[gidx] = z;
  }
  __syncthreads();
  {
    float z = A.fb0[ti][tid];
    for (int d = 0; d < 512; ++d) z = fmaf(G[d], A.fW0[ti][d * 64 + tid], z);
    y0s[tid] = htanh(z);
  }
  __syncthreads();
  {
    float z = A.fb1[ti][tid];
    for (int d = 0; d < 64; ++d) z = fmaf(y0s[d], A.fW1[ti][d * 64 + tid], z);
    y1s[tid] = y0s[tid] + htanh(z);
  }
  __syncthreads();
  red[tid] = y1s[tid] * A.fW2[ti][tid];
  __syncthreads();
  for (int s = 32; s > 0; s >>= 1) { if (tid < s) red[tid] += red[tid + s]; __syncthreads(); }
  if (tid == 0) A.EA[n] = red[0] + A.fb2[ti][0] + ((ti == 0) ? -93.0f : -187.0f);
}

// Kernel 5: fixed-order deterministic energy reduction
__global__ __launch_bounds__(64) void k5_red(const float* __restrict__ ea, float* __restrict__ out) {
  __shared__ float red[64];
  int tid = threadIdx.x;
  red[tid] = ea[tid] + ea[tid + 64] + ea[tid + 128];
  __syncthreads();
  for (int s = 32; s > 0; s >>= 1) { if (tid < s) red[tid] += red[tid + s]; __syncthreads(); }
  if (tid == 0) out[0] = red[0];
}

// ===========================================================================
extern "C" void kernel_launch(void* const* d_in, const int* in_sizes, int n_in,
                              void* d_out, int out_size, void* d_ws, size_t ws_size,
                              hipStream_t stream) {
  (void)in_sizes; (void)n_in; (void)out_size; (void)ws_size;
  auto F = [&](int i) { return reinterpret_cast<const float*>(d_in[i]); };
  float* ws = reinterpret_cast<float*>(d_ws);
  float* T4 = ws + WS_T4;
  float* F1 = ws + WS_F1;
  float* F3 = ws + WS_F3;
  float* TX = ws + WS_TX;
  float* EA = ws + WS_EA;

  K1Args a1; a1.coord = F(0); a1.box = F(1); a1.T4 = T4;
  for (int i = 0; i < 2; ++i)
    for (int j = 0; j < 2; ++j)
      for (int k = 0; k < 2; ++k) {
        int idx = (i * 2 + j) * 2 + k, b = 3 + idx * 4;
        a1.ijW0[idx] = F(b); a1.ijb0[idx] = F(b + 1);
        a1.ijW1[idx] = F(b + 2); a1.ijb1[idx] = F(b + 3);
      }

  K2Args a2; a2.T4 = T4; a2.F1 = F1; a2.F3 = F3;
  for (int i = 0; i < 2; ++i) {
    a2.L1[i] = F(95 + i); a2.L3[i] = F(97 + i);
    a2.N1[i] = F(99 + i); a2.N2[i] = F(101 + i);
  }

  K3Args a3; a3.coord = F(0); a3.box = F(1); a3.F1 = F1; a3.F3 = F3; a3.TX = TX;
  for (int ij = 0; ij < 4; ++ij) {
    int b = 59 + ij * 6;
    a3.erW0[ij] = F(b);     a3.erb0[ij] = F(b + 1); a3.erW1[ij] = F(b + 2);
    a3.erb1[ij] = F(b + 3); a3.erW2[ij] = F(b + 4); a3.erb2[ij] = F(b + 5);
    int c = 35 + ij * 6;
    a3.mpb0[ij] = F(c);     a3.mpW1[ij] = F(c + 1); a3.mpb1[ij] = F(c + 2);
    a3.mpW2[ij] = F(c + 3); a3.mpb2[ij] = F(c + 4); a3.mpdt[ij] = F(c + 5);
  }

  K4Args a4; a4.TX = TX; a4.Tbias = F(2); a4.EA = EA;
  for (int i = 0; i < 2; ++i) {
    int b = 83 + i * 6;
    a4.fW0[i] = F(b);     a4.fb0[i] = F(b + 1); a4.fW1[i] = F(b + 2);
    a4.fb1[i] = F(b + 3); a4.fW2[i] = F(b + 4); a4.fb2[i] = F(b + 5);
  }

  k1_t4 <<<NATOM, 128, 0, stream>>>(a1);
  k2_proj<<<NATOM,  32, 0, stream>>>(a2);
  k3_mp  <<<NATOM,  64, 0, stream>>>(a3);
  k4_fit <<<NATOM,  64, 0, stream>>>(a4);
  k5_red <<<1,      64, 0, stream>>>(EA, reinterpret_cast<float*>(d_out));
}
